// MoE_67242007986678
// MI455X (gfx1250) — matrix operand order repaired and checked by reading, not compiled, not run on hardware
//
#include <hip/hip_runtime.h>

// ---------------------------------------------------------------------------
// MoE (shared expert + 8 routed experts, top-2) for gfx1250 (MI455X).
//
// Pipeline:
//   1) one-time fp32->bf16 conversion of x, and convert+transpose of all
//      weight matrices into [M x K] bf16 (workspace) -- done once per launch
//   2) bf16 WMMA GEMMs (v_wmma_f32_16x16x32_bf16, f32 accum) whose LDS tile
//      staging is a pure 16B copy via GLOBAL_LOAD_ASYNC_TO_LDS_B128
//      (ASYNCcnt) -- no per-tile conversion VALU in the hot loop.
// ---------------------------------------------------------------------------

#define E_EXPERTS 8
#define DIMD      2048   // model dim
#define FDIM      1024   // routed expert intermediate
#define FSDIM     2048   // shared expert intermediate
#define NTOK      2048   // B*T

#define BM   128
#define BN   128
#define BK   32
#define LSTR 40          // LDS row stride in ushort (32 + 8 pad), keeps 16B align

typedef __attribute__((ext_vector_type(16))) __bf16          v16bf;
typedef __attribute__((ext_vector_type(8)))  float           v8f;
typedef __attribute__((ext_vector_type(8)))  unsigned short  us8;
typedef __attribute__((ext_vector_type(4)))  int             v4i;
typedef unsigned short ushort_t;

union FragU {
    v16bf v;
    us8   h[2];
};

__device__ __forceinline__ unsigned short f2bf(float f) {
    // round-to-nearest-even fp32 -> bf16 (integer path; deterministic)
    unsigned u = __builtin_bit_cast(unsigned, f);
    u += 0x7FFFu + ((u >> 16) & 1u);
    return (unsigned short)(u >> 16);
}

__device__ __forceinline__ float silu_f(float g) {
    return g / (1.0f + __expf(-g));
}

// 16-byte asynchronous global -> LDS copy (gfx1250 async path, ASYNCcnt).
__device__ __forceinline__ void async_cp16(const void* g, const void* l) {
#if __has_builtin(__builtin_amdgcn_global_load_async_to_lds_b128)
    // signature (probe-confirmed via diagnostic): (v4i global* src, v4i lds* dst, imm, imm)
    __builtin_amdgcn_global_load_async_to_lds_b128(
        (__attribute__((address_space(1))) v4i*)(size_t)g,
        (__attribute__((address_space(3))) v4i*)(unsigned)(size_t)l,
        0, 0);
#else
    const unsigned lds_off = (unsigned)(size_t)l;   // low 32 bits == LDS offset
    asm volatile("global_load_async_to_lds_b128 %0, %1, off"
                 :: "v"(lds_off), "v"(g) : "memory");
#endif
}

__device__ __forceinline__ void async_wait0() {
    asm volatile("s_wait_asynccnt 0" ::: "memory");
}

// ---------------------------------------------------------------------------
// Router: probs = softmax(x @ router_w), top-2 -> dense combine[N, E]
// One wave (32 lanes, wave32) per token.
// ---------------------------------------------------------------------------
__global__ __launch_bounds__(256) void moe_router(
    const float* __restrict__ x,        // [N, D]
    const float* __restrict__ rw,       // [D, E]
    float* __restrict__ combine)        // [N, E]
{
    const int wave = threadIdx.x >> 5;
    const int lane = threadIdx.x & 31;
    const int n    = blockIdx.x * 8 + wave;

    const float* xr = x + (size_t)n * DIMD;
    float acc[E_EXPERTS];
#pragma unroll
    for (int e = 0; e < E_EXPERTS; ++e) acc[e] = 0.0f;

    for (int d = lane; d < DIMD; d += 32) {
        const float xv = xr[d];
        const float* w = rw + (size_t)d * E_EXPERTS;
#pragma unroll
        for (int e = 0; e < E_EXPERTS; ++e) acc[e] += xv * w[e];
    }
#pragma unroll
    for (int off = 16; off > 0; off >>= 1) {
#pragma unroll
        for (int e = 0; e < E_EXPERTS; ++e)
            acc[e] += __shfl_xor(acc[e], off, 32);
    }
    if (lane == 0) {
        float m = acc[0];
#pragma unroll
        for (int e = 1; e < E_EXPERTS; ++e) m = fmaxf(m, acc[e]);
        float p[E_EXPERTS];
        float s = 0.0f;
#pragma unroll
        for (int e = 0; e < E_EXPERTS; ++e) { p[e] = __expf(acc[e] - m); s += p[e]; }
        const float inv = 1.0f / s;
#pragma unroll
        for (int e = 0; e < E_EXPERTS; ++e) p[e] *= inv;

        int i0 = 0;
#pragma unroll
        for (int e = 1; e < E_EXPERTS; ++e) if (p[e] > p[i0]) i0 = e;
        int i1 = (i0 == 0) ? 1 : 0;
#pragma unroll
        for (int e = 0; e < E_EXPERTS; ++e)
            if (e != i0 && p[e] > p[i1]) i1 = e;

        float* outr = combine + (size_t)n * E_EXPERTS;
#pragma unroll
        for (int e = 0; e < E_EXPERTS; ++e) outr[e] = 0.0f;
        outr[i0] = p[i0];
        outr[i1] = p[i1];
    }
}

// ---------------------------------------------------------------------------
// One-time fp32 -> bf16 conversion (no transpose).
// ---------------------------------------------------------------------------
__global__ __launch_bounds__(256) void cvt_bf16(
    const float* __restrict__ in, ushort_t* __restrict__ out)
{
    const size_t idx = (size_t)blockIdx.x * 256 + threadIdx.x;
    const float4 v = *(const float4*)(in + idx * 4);
    unsigned long long packed =
        (unsigned long long)f2bf(v.x)
        | ((unsigned long long)f2bf(v.y) << 16)
        | ((unsigned long long)f2bf(v.z) << 32)
        | ((unsigned long long)f2bf(v.w) << 48);
    *(unsigned long long*)(out + idx * 4) = packed;
}

// ---------------------------------------------------------------------------
// One-time convert + transpose: in f32 [R x C] -> out bf16 [C x R].
// 32x32 LDS tile, 256 threads.
// ---------------------------------------------------------------------------
__global__ __launch_bounds__(256) void cvt_tr_bf16(
    const float* __restrict__ in, ushort_t* __restrict__ out, int R, int C)
{
    __shared__ ushort_t tile[32 * 33];
    const int tx = threadIdx.x & 31;
    const int ty = threadIdx.x >> 5;
    const int r0 = blockIdx.y * 32;
    const int c0 = blockIdx.x * 32;
#pragma unroll
    for (int j = 0; j < 4; ++j) {
        const int r = ty + j * 8;
        tile[tx * 33 + r] = f2bf(in[(size_t)(r0 + r) * C + c0 + tx]);
    }
    __syncthreads();
#pragma unroll
    for (int j = 0; j < 4; ++j) {
        const int c = ty + j * 8;
        out[(size_t)(c0 + c) * R + r0 + tx] = tile[c * 33 + tx];
    }
}

// ---------------------------------------------------------------------------
// Fused gate/up dual GEMM + SwiGLU epilogue (all-bf16 operands):
//   H[N,M](bf16) = silu(A @ Wg^T') * (A @ Wu^T')
//   A: [N,K] bf16 row-major;  Bg/Bu: [M,K] bf16 (pre-transposed weights)
// 128x128x32 tiles; 8 waves (4 row x 2 col); each wave 2x4 WMMA tiles.
// LDS staging via async 16B copies.
// ---------------------------------------------------------------------------
__global__ __launch_bounds__(256) void gemm_gateup(
    const ushort_t* __restrict__ A,
    const ushort_t* __restrict__ Bg,
    const ushort_t* __restrict__ Bu,
    ushort_t* __restrict__ H,
    int K, int M)
{
    __shared__ __attribute__((aligned(16))) ushort_t As [BM * LSTR];
    __shared__ __attribute__((aligned(16))) ushort_t Bgs[BN * LSTR];
    __shared__ __attribute__((aligned(16))) ushort_t Bus[BN * LSTR];

    const int t    = threadIdx.x;
    const int lane = t & 31;
    const int w    = t >> 5;
    const int wr   = w & 3;        // 0..3  -> 32-row strip
    const int wc   = w >> 2;       // 0..1  -> 64-col strip
    const int g    = lane >> 4;    // half-wave group
    const int ln   = lane & 15;

    const int bm0 = blockIdx.y * BM;
    const int bn0 = blockIdx.x * BN;

    v8f cg[2][4], cu[2][4];
#pragma unroll
    for (int mt = 0; mt < 2; ++mt)
#pragma unroll
        for (int nt = 0; nt < 4; ++nt)
#pragma unroll
            for (int v = 0; v < 8; ++v) { cg[mt][nt][v] = 0.0f; cu[mt][nt][v] = 0.0f; }

    const int nk = K / BK;
    for (int kt = 0; kt < nk; ++kt) {
        // ---- async staging: 3 tiles x 512 sixteen-byte chunks ----
#pragma unroll
        for (int i = 0; i < 2; ++i) {
            const int idx = t + i * 256;
            const int row = idx >> 2;
            const int cv  = idx & 3;
            const size_t gofs = (size_t)row * K + kt * BK + cv * 8;
            const int    lofs = row * LSTR + cv * 8;
            async_cp16(A  + (size_t)bm0 * K + gofs, As  + lofs);
            async_cp16(Bg + (size_t)bn0 * K + gofs, Bgs + lofs);
            async_cp16(Bu + (size_t)bn0 * K + gofs, Bus + lofs);
        }
        if (kt + 1 < nk) {
            __builtin_prefetch(A + (size_t)(bm0 + (t >> 1)) * K + (kt + 1) * BK, 0, 1);
        }
        async_wait0();
        __syncthreads();

        // ---- per-wave fragments + WMMAs ----
        FragU a[2];
#pragma unroll
        for (int mt = 0; mt < 2; ++mt) {
            const ushort_t* p = As + (wr * 32 + mt * 16 + ln) * LSTR;
            a[mt].h[0] = *(const us8*)(p + g * 16);         // K = 16g .. 16g+7
            a[mt].h[1] = *(const us8*)(p + g * 16 + 8);     // K = 16g+8 .. 16g+15
        }
#pragma unroll
        for (int nt = 0; nt < 4; ++nt) {
            const ushort_t* pg = Bgs + (wc * 64 + nt * 16 + ln) * LSTR + g * 16;
            const ushort_t* pu = Bus + (wc * 64 + nt * 16 + ln) * LSTR + g * 16;
            FragU bgf, buf;
            bgf.h[0] = *(const us8*)pg;   bgf.h[1] = *(const us8*)(pg + 8);
            buf.h[0] = *(const us8*)pu;   buf.h[1] = *(const us8*)(pu + 8);
#pragma unroll
            for (int mt = 0; mt < 2; ++mt) {
                cg[mt][nt] = __builtin_amdgcn_wmma_f32_16x16x32_bf16(
                    false, a[mt].v, false, bgf.v, (short)0, cg[mt][nt], false, false);
                cu[mt][nt] = __builtin_amdgcn_wmma_f32_16x16x32_bf16(
                    false, a[mt].v, false, buf.v, (short)0, cu[mt][nt], false, false);
            }
        }
        __syncthreads();
    }

    // ---- epilogue: H = bf16(silu(g) * u) ----
#pragma unroll
    for (int mt = 0; mt < 2; ++mt) {
#pragma unroll
        for (int nt = 0; nt < 4; ++nt) {
            const int c = bn0 + wc * 64 + nt * 16 + ln;
#pragma unroll
            for (int v = 0; v < 8; ++v) {
                const int r = bm0 + wr * 32 + mt * 16 + g * 8 + v;
                H[(size_t)r * M + c] = f2bf(silu_f(cg[mt][nt][v]) * cu[mt][nt][v]);
            }
        }
    }
}

// ---------------------------------------------------------------------------
// Down projection (all-bf16 operands):
//   out[N,D](f32) (+)= scale_row * (H @ Wd^T'),  H: [N,K] bf16, Bt: [D,K] bf16
// scale_row = combine[n*E + e] when combine != nullptr, else 1.
// ---------------------------------------------------------------------------
__global__ __launch_bounds__(256) void gemm_down(
    const ushort_t* __restrict__ Hin,
    const ushort_t* __restrict__ Bt,
    float* __restrict__ out,
    const float* __restrict__ combine,   // may be nullptr
    int e, int K, int accumulate)
{
    __shared__ __attribute__((aligned(16))) ushort_t As[BM * LSTR];
    __shared__ __attribute__((aligned(16))) ushort_t Bs[BN * LSTR];

    const int t    = threadIdx.x;
    const int lane = t & 31;
    const int w    = t >> 5;
    const int wr   = w & 3;
    const int wc   = w >> 2;
    const int g    = lane >> 4;
    const int ln   = lane & 15;

    const int bm0 = blockIdx.y * BM;
    const int bn0 = blockIdx.x * BN;

    v8f cc[2][4];
#pragma unroll
    for (int mt = 0; mt < 2; ++mt)
#pragma unroll
        for (int nt = 0; nt < 4; ++nt)
#pragma unroll
            for (int v = 0; v < 8; ++v) cc[mt][nt][v] = 0.0f;

    const int nk = K / BK;
    for (int kt = 0; kt < nk; ++kt) {
#pragma unroll
        for (int i = 0; i < 2; ++i) {
            const int idx = t + i * 256;
            const int row = idx >> 2;
            const int cv  = idx & 3;
            const size_t gofs = (size_t)row * K + kt * BK + cv * 8;
            const int    lofs = row * LSTR + cv * 8;
            async_cp16(Hin + (size_t)bm0 * K + gofs, As + lofs);
            async_cp16(Bt  + (size_t)bn0 * K + gofs, Bs + lofs);
        }
        if (kt + 1 < nk) {
            __builtin_prefetch(Hin + (size_t)(bm0 + (t >> 1)) * K + (kt + 1) * BK, 0, 1);
        }
        async_wait0();
        __syncthreads();

        FragU a[2];
#pragma unroll
        for (int mt = 0; mt < 2; ++mt) {
            const ushort_t* p = As + (wr * 32 + mt * 16 + ln) * LSTR;
            a[mt].h[0] = *(const us8*)(p + g * 16);
            a[mt].h[1] = *(const us8*)(p + g * 16 + 8);
        }
#pragma unroll
        for (int nt = 0; nt < 4; ++nt) {
            const ushort_t* pb = Bs + (wc * 64 + nt * 16 + ln) * LSTR + g * 16;
            FragU bf;
            bf.h[0] = *(const us8*)pb;  bf.h[1] = *(const us8*)(pb + 8);
#pragma unroll
            for (int mt = 0; mt < 2; ++mt) {
                cc[mt][nt] = __builtin_amdgcn_wmma_f32_16x16x32_bf16(
                    false, a[mt].v, false, bf.v, (short)0, cc[mt][nt], false, false);
            }
        }
        __syncthreads();
    }

#pragma unroll
    for (int mt = 0; mt < 2; ++mt) {
#pragma unroll
        for (int nt = 0; nt < 4; ++nt) {
            const int c = bn0 + wc * 64 + nt * 16 + ln;
#pragma unroll
            for (int v = 0; v < 8; ++v) {
                const int r = bm0 + wr * 32 + mt * 16 + g * 8 + v;
                const float sc = combine ? combine[(size_t)r * E_EXPERTS + e] : 1.0f;
                const size_t o = (size_t)r * DIMD + c;
                const float prev = accumulate ? out[o] : 0.0f;
                out[o] = prev + sc * cc[mt][nt][v];
            }
        }
    }
}

// ---------------------------------------------------------------------------
// Host-side launch sequence (graph-capture safe: only kernel launches).
// ---------------------------------------------------------------------------
extern "C" void kernel_launch(void* const* d_in, const int* in_sizes, int n_in,
                              void* d_out, int out_size, void* d_ws, size_t ws_size,
                              hipStream_t stream) {
    (void)in_sizes; (void)n_in; (void)out_size; (void)ws_size;

    const float* x  = (const float*)d_in[0];   // [N, D]
    const float* rw = (const float*)d_in[1];   // [D, E]
    const float* sg = (const float*)d_in[2];   // [D, FS]
    const float* su = (const float*)d_in[3];   // [D, FS]
    const float* sd = (const float*)d_in[4];   // [FS, D]
    const float* wg = (const float*)d_in[5];   // [E, D, F]
    const float* wu = (const float*)d_in[6];   // [E, D, F]
    const float* wd = (const float*)d_in[7];   // [E, F, D]
    float* out = (float*)d_out;                // [N, D]

    // workspace layout (256B-aligned slots)
    char* ws = (char*)d_ws;
    const size_t SZ_COMB = ((size_t)NTOK * E_EXPERTS * sizeof(float) + 255) & ~(size_t)255;
    const size_t SZ_BF   = ((size_t)2048 * 2048 * sizeof(ushort_t) + 255) & ~(size_t)255; // 8MB slots
    float*    combine = (float*)ws;
    ushort_t* xbf     = (ushort_t*)(ws + SZ_COMB);
    ushort_t* Hbf     = (ushort_t*)(ws + SZ_COMB + SZ_BF);
    ushort_t* BgT     = (ushort_t*)(ws + SZ_COMB + 2 * SZ_BF);
    ushort_t* BuT     = (ushort_t*)(ws + SZ_COMB + 3 * SZ_BF);
    ushort_t* BdT     = (ushort_t*)(ws + SZ_COMB + 4 * SZ_BF);

    const dim3 blk(256);

    // 1) router -> combine weights
    moe_router<<<NTOK / 8, blk, 0, stream>>>(x, rw, combine);

    // 2) x -> bf16 once
    cvt_bf16<<<(NTOK * DIMD) / (4 * 256), blk, 0, stream>>>(x, xbf);

    // 3) shared expert
    cvt_tr_bf16<<<dim3(FSDIM / 32, DIMD / 32), blk, 0, stream>>>(sg, BgT, DIMD, FSDIM);
    cvt_tr_bf16<<<dim3(FSDIM / 32, DIMD / 32), blk, 0, stream>>>(su, BuT, DIMD, FSDIM);
    gemm_gateup<<<dim3(FSDIM / BN, NTOK / BM), blk, 0, stream>>>(xbf, BgT, BuT, Hbf, DIMD, FSDIM);
    cvt_tr_bf16<<<dim3(DIMD / 32, FSDIM / 32), blk, 0, stream>>>(sd, BdT, FSDIM, DIMD);
    gemm_down<<<dim3(DIMD / BN, NTOK / BM), blk, 0, stream>>>(Hbf, BdT, out, nullptr, 0, FSDIM, 0);

    // 4) routed experts: out += combine[:,e] * (silu(x@Wg_e)*(x@Wu_e)) @ Wd_e
    for (int e = 0; e < E_EXPERTS; ++e) {
        const float* wge = wg + (size_t)e * DIMD * FDIM;
        const float* wue = wu + (size_t)e * DIMD * FDIM;
        const float* wde = wd + (size_t)e * FDIM * DIMD;
        cvt_tr_bf16<<<dim3(FDIM / 32, DIMD / 32), blk, 0, stream>>>(wge, BgT, DIMD, FDIM);
        cvt_tr_bf16<<<dim3(FDIM / 32, DIMD / 32), blk, 0, stream>>>(wue, BuT, DIMD, FDIM);
        gemm_gateup<<<dim3(FDIM / BN, NTOK / BM), blk, 0, stream>>>(xbf, BgT, BuT, Hbf, DIMD, FDIM);
        cvt_tr_bf16<<<dim3(DIMD / 32, FDIM / 32), blk, 0, stream>>>(wde, BdT, FDIM, DIMD);
        gemm_down<<<dim3(DIMD / BN, NTOK / BM), blk, 0, stream>>>(Hbf, BdT, out, combine, e, FDIM, 1);
    }
}
